// TransformerEncoderLayer_57947698757781
// MI455X (gfx1250) — compile-verified, matrix operand order checked by reading, not run on hardware
//
#include <hip/hip_runtime.h>

// ---------------------------------------------------------------------------
// Types for CDNA5 WMMA (wave32): V_WMMA_F32_16X16X32_F16
// ---------------------------------------------------------------------------
typedef _Float16 v16h __attribute__((ext_vector_type(16)));
typedef _Float16 v8h  __attribute__((ext_vector_type(8)));
typedef float    v8f  __attribute__((ext_vector_type(8)));

#define TLEN 2048
#define BATCH 8
#define HDIM 256
#define DDIM 768
#define G3H  768   // 3*H
#define ASTR 520   // f16 A-staging row stride (260 dwords = 4 mod 64 banks)
#define GSTR 772   // f32 gh/gi scratch row stride (772 = 4 mod 64 banks)

__device__ __forceinline__ v8f wmma16x16x32(v16h a, v16h b, v8f c) {
  return __builtin_amdgcn_wmma_f32_16x16x32_f16(
      /*neg_a=*/false, a, /*neg_b=*/false, b,
      /*c_mod=*/(short)0, c, /*reuse_a=*/false, /*reuse_b=*/false);
}

// Load a 16-half fragment for the 16x16x32 f16 WMMA A/B operand.
// ISA layout (CDNA5 ISA 7.12.2): lane group g = lane>>4 selects K-halves.
//   VGPR0..3 hold K = g*8 .. g*8+7 ; VGPR4..7 hold K = 16+g*8 .. 16+g*8+7
// => two contiguous 16-byte chunks at (kbase + g*8) and (kbase + g*8 + 16).
// 'row' is the matrix row (A: m = lane&15) or weight row == output col (B).
template <typename PT>
__device__ __forceinline__ v16h load_frag(const PT* base, int row, int stride,
                                          int kbase, int g) {
  const _Float16* p = (const _Float16*)base + (size_t)row * stride + kbase + g * 8;
  v8h lo = *(const v8h*)(p);
  v8h hi = *(const v8h*)(p + 16);
  return __builtin_shufflevector(lo, hi, 0, 1, 2, 3, 4, 5, 6, 7,
                                 8, 9, 10, 11, 12, 13, 14, 15);
}

__device__ __forceinline__ float sigmoidf_(float x) {
  return 1.0f / (1.0f + __expf(-x));
}

// ---------------------------------------------------------------------------
// Elementwise conversion kernels
// ---------------------------------------------------------------------------
__global__ __launch_bounds__(256) void cvt_f32_to_f16(
    const float* __restrict__ in, _Float16* __restrict__ out, size_t n) {
  size_t i = (size_t)blockIdx.x * 256 + threadIdx.x;
  if (i < n) out[i] = (_Float16)in[i];
}

// x: (B,T,D) f32  ->  x2h: row = t*8+b, col = d (f16), so a 16-row WMMA tile
// covers two timesteps x all 8 batches, and the scan reads 8 contiguous rows.
__global__ __launch_bounds__(256) void xpose_to_f16(
    const float* __restrict__ x, _Float16* __restrict__ out) {
  size_t idx = (size_t)blockIdx.x * 256 + threadIdx.x;
  const size_t n = (size_t)BATCH * TLEN * DDIM;
  if (idx >= n) return;
  int d = (int)(idx % DDIM);
  size_t bt = idx / DDIM;
  int t = (int)(bt % TLEN);
  int b = (int)(bt / TLEN);
  out[((size_t)t * BATCH + b) * DDIM + d] = (_Float16)x[idx];
}

// ---------------------------------------------------------------------------
// Batched GEMM (N fixed at 768): C[m][n] = sum_k A[m][k]*W[n][k] + bias[n]
// A: f16 [M x K] row-major (shared across batch), W: f16 [N x K] row-major,
// C: f32 [M x N].  Block = 256 threads = 8 wave32s; each wave owns 6 N-tiles
// so one A fragment feeds 6 WMMAs (6 independent accumulation chains).
// ---------------------------------------------------------------------------
__global__ __launch_bounds__(256) void gemm_wmma_f16(
    const _Float16* __restrict__ A, const _Float16* __restrict__ W,
    const float* __restrict__ bias, float* __restrict__ C,
    int K, int N, long wStride, long bStride, long cStride) {
  const int lane = threadIdx.x & 31;
  const int wave = threadIdx.x >> 5;
  const int g = lane >> 4, l15 = lane & 15;
  const int mtile = blockIdx.x;
  const int bat = blockIdx.z;

  const _Float16* Wb = W + (size_t)bat * wStride;
  const float* bb = bias + (size_t)bat * bStride;
  float* Cb = C + (size_t)bat * cStride;

  const int arow = mtile * 16 + l15;

  v8f acc[6] = {};
  for (int kb = 0; kb < K; kb += 32) {
    v16h af = load_frag(A, arow, K, kb, g);
#pragma unroll
    for (int tq = 0; tq < 6; ++tq) {
      const int wrow = (wave * 6 + tq) * 16 + l15;
      v16h bf = load_frag(Wb, wrow, K, kb, g);
      acc[tq] = wmma16x16x32(af, bf, acc[tq]);
    }
  }
#pragma unroll
  for (int tq = 0; tq < 6; ++tq) {
    const int n = (wave * 6 + tq) * 16 + l15;
    const float bv = bb[n];
#pragma unroll
    for (int v = 0; v < 8; ++v) {
      int m = mtile * 16 + v + 8 * g;  // C/D layout: VGPR v, lanes>=16 -> M+8
      Cb[(size_t)m * N + n] = acc[tq][v] + bv;
    }
  }
}

// ---------------------------------------------------------------------------
// Persistent per-span scan. grid = 3 blocks (span 1/3/5), block = 512 threads
// = 16 wave32s. All recurrent state lives in LDS (246.5 KB dynamic):
//   sH  : h state   [para(2)][layer(2)][16][256] f32   (rows 8..15 = padding)
//   sHP : skip accum (same shape) f32
//   sGH : gh scratch [16][GSTR] f32
//   sGI : gi scratch [16][GSTR] f32
//   sA  : f16 A-operand staging [16][ASTR] (cols 0..511 used)
// ---------------------------------------------------------------------------
__global__ __launch_bounds__(512) void span_scan(
    const _Float16* __restrict__ whh0h, const _Float16* __restrict__ wih1h,
    const _Float16* __restrict__ whh1h, const _Float16* __restrict__ gateWh,
    const float* __restrict__ bhh0, const float* __restrict__ bih1,
    const float* __restrict__ bhh1, const float* __restrict__ gate_b,
    const float* __restrict__ h0, const float* __restrict__ gi0,
    float* __restrict__ cat) {
  extern __shared__ char smem_raw[];
  float* sH = (float*)smem_raw;                 // 16384 f32
  float* sHP = sH + 16384;                      // 16384 f32
  float* sGH = sHP + 16384;                     // 16*GSTR f32
  float* sGI = sGH + 16 * GSTR;                 // 16*GSTR f32
  _Float16* sA = (_Float16*)(sGI + 16 * GSTR);  // 16*ASTR f16

  const int tid = threadIdx.x;
  const int lane = tid & 31;
  const int wave = tid >> 5;
  const int g = lane >> 4, l15 = lane & 15;
  const int sidx = blockIdx.x;
  const int span = (sidx == 0) ? 1 : (sidx == 1 ? 3 : 5);

  for (int i = 0; i < TLEN; ++i) {
    // ---- per-step uniform flags: permuted index -> original t --------------
    int grp = 0, start = 0;
    for (;;) {
      int len = (TLEN - grp + span - 1) / span;
      if (i < start + len) break;
      start += len;
      ++grp;
    }
    const int off = i - start;
    const int t = grp + off * span;
    const bool reset = (off == 0);
    const bool skipf = (((i + 1) % 6) == 0);
    const bool poolf = (((i + 1) % 4) != 0);

    // ---- periodic reset of h (NOT hp); i==0 also initializes hp ------------
    if (reset) {
      for (int idx = tid; idx < 16384; idx += 512) {
        int p = idx >> 13, l = (idx >> 12) & 1;
        int b = (idx >> 8) & 15, h = idx & 255;
        float v = (b < 8) ? h0[(((p * 2 + l) * BATCH) + b) * HDIM + h] : 0.0f;
        sH[idx] = v;
        if (i == 0) sHP[idx] = v;
      }
    }
    __syncthreads();

    // ---- two 2-layer GRUs (para p = 0,1); weight index j = 2*sidx + p ------
    for (int p = 0; p < 2; ++p) {
      const int j = sidx * 2 + p;
      // stage layer-0 h as f16 A operand
      for (int idx = tid; idx < 4096; idx += 512) {
        int b = idx >> 8, h = idx & 255;
        sA[b * ASTR + h] = (_Float16)sH[((p * 2 + 0) * 16 + b) * HDIM + h];
      }
      __syncthreads();

      // gh0 = h0_prev @ Whh0[j]^T : (16x256)x(256x768), 48 N-tiles / 16 waves
      {
        const _Float16* W0 = whh0h + (size_t)j * G3H * HDIM;
        v16h afr[8];
#pragma unroll
        for (int kb = 0; kb < 8; ++kb)
          afr[kb] = load_frag(sA, l15, ASTR, kb * 32, g);
        for (int nt = wave; nt < 48; nt += 16) {
          v8f acc = {};
#pragma unroll
          for (int kb = 0; kb < 8; ++kb) {
            v16h bf = load_frag(W0, nt * 16 + l15, HDIM, kb * 32, g);
            acc = wmma16x16x32(afr[kb], bf, acc);
          }
          const int n = nt * 16 + l15;
#pragma unroll
          for (int v = 0; v < 8; ++v) sGH[(v + 8 * g) * GSTR + n] = acc[v];
        }
      }
      __syncthreads();

      // layer-0 elementwise (gi0 precomputed incl. bih0) + stage layer-1 A
      for (int idx = tid; idx < 4096; idx += 512) {
        int b = idx >> 8, h = idx & 255;
        float gr = 0.f, gz = 0.f, gn = 0.f;
        if (b < 8) {
          const float* gi = gi0 + (((size_t)j * TLEN + t) * BATCH + b) * G3H;
          gr = gi[h];
          gz = gi[256 + h];
          gn = gi[512 + h];
        }
        const float* bh = bhh0 + j * G3H;
        float hr = sGH[b * GSTR + h] + bh[h];
        float hz = sGH[b * GSTR + 256 + h] + bh[256 + h];
        float hn = sGH[b * GSTR + 512 + h] + bh[512 + h];
        float r = sigmoidf_(gr + hr);
        float z = sigmoidf_(gz + hz);
        float nst = tanhf(gn + r * hn);
        float hold = sH[((p * 2 + 0) * 16 + b) * HDIM + h];
        float hnew = (1.0f - z) * nst + z * hold;
        sH[((p * 2 + 0) * 16 + b) * HDIM + h] = hnew;
        sA[b * ASTR + h] = (_Float16)hnew;  // layer-1 input x
        sA[b * ASTR + 256 + h] =
            (_Float16)sH[((p * 2 + 1) * 16 + b) * HDIM + h];  // layer-1 h
      }
      __syncthreads();

      // layer 1: pass0: gi1 = hnew0 @ Wih1^T ; pass1: gh1 = h1_prev @ Whh1^T
      for (int pass = 0; pass < 2; ++pass) {
        const _Float16* Wp = pass ? (whh1h + (size_t)j * G3H * HDIM)
                                  : (wih1h + (size_t)j * G3H * HDIM);
        float* outb = pass ? sGH : sGI;
        const int abase = pass ? 256 : 0;
        v16h afr[8];
#pragma unroll
        for (int kb = 0; kb < 8; ++kb)
          afr[kb] = load_frag(sA, l15, ASTR, abase + kb * 32, g);
        for (int nt = wave; nt < 48; nt += 16) {
          v8f acc = {};
#pragma unroll
          for (int kb = 0; kb < 8; ++kb) {
            v16h bf = load_frag(Wp, nt * 16 + l15, HDIM, kb * 32, g);
            acc = wmma16x16x32(afr[kb], bf, acc);
          }
          const int n = nt * 16 + l15;
#pragma unroll
          for (int v = 0; v < 8; ++v) outb[(v + 8 * g) * GSTR + n] = acc[v];
        }
      }
      __syncthreads();

      // layer-1 elementwise
      for (int idx = tid; idx < 4096; idx += 512) {
        int b = idx >> 8, h = idx & 255;
        const float* bi = bih1 + j * G3H;
        const float* bh = bhh1 + j * G3H;
        float ir = sGI[b * GSTR + h] + bi[h];
        float iz = sGI[b * GSTR + 256 + h] + bi[256 + h];
        float inn = sGI[b * GSTR + 512 + h] + bi[512 + h];
        float hr = sGH[b * GSTR + h] + bh[h];
        float hz = sGH[b * GSTR + 256 + h] + bh[256 + h];
        float hn = sGH[b * GSTR + 512 + h] + bh[512 + h];
        float r = sigmoidf_(ir + hr);
        float z = sigmoidf_(iz + hz);
        float nst = tanhf(inn + r * hn);
        float hold = sH[((p * 2 + 1) * 16 + b) * HDIM + h];
        sH[((p * 2 + 1) * 16 + b) * HDIM + h] = (1.0f - z) * nst + z * hold;
      }
      __syncthreads();
    }  // para loop

    // ---- skip-span accumulation every 6th step ----------------------------
    if (skipf) {
      for (int idx = tid; idx < 16384; idx += 512) {
        float hs = sH[idx] + sHP[idx];
        sH[idx] = hs;
        sHP[idx] = hs;
      }
    }
    __syncthreads();

    // ---- gated fusion per layer: g = sigmoid([h1|h2] @ gateW^T + b) --------
    for (int l = 0; l < 2; ++l) {
      for (int idx = tid; idx < 4096; idx += 512) {
        int b = idx >> 8, h = idx & 255;
        sA[b * ASTR + h] = (_Float16)sH[((0 * 2 + l) * 16 + b) * HDIM + h];
        sA[b * ASTR + 256 + h] =
            (_Float16)sH[((1 * 2 + l) * 16 + b) * HDIM + h];
      }
      __syncthreads();
      {  // (16x512)x(512x256): 16 N-tiles, one per wave, K = 16 steps of 32
        const int nt = wave;
        v8f acc = {};
#pragma unroll
        for (int kb = 0; kb < 16; ++kb) {
          v16h af = load_frag(sA, l15, ASTR, kb * 32, g);
          v16h bf = load_frag(gateWh, nt * 16 + l15, 512, kb * 32, g);
          acc = wmma16x16x32(af, bf, acc);
        }
        const int n = nt * 16 + l15;
#pragma unroll
        for (int v = 0; v < 8; ++v) sGH[(v + 8 * g) * GSTR + n] = acc[v];
      }
      __syncthreads();
      if (poolf) {
        for (int idx = tid; idx < 4096; idx += 512) {
          int b = idx >> 8, h = idx & 255;
          float gv = sigmoidf_(sGH[b * GSTR + h] + gate_b[h]);
          float h1v = sH[((0 * 2 + l) * 16 + b) * HDIM + h];
          float h2v = sH[((1 * 2 + l) * 16 + b) * HDIM + h];
          sH[((0 * 2 + l) * 16 + b) * HDIM + h] =
              gv * h1v + (1.0f - gv) * h2v;
        }
      }
      __syncthreads();
    }

    // ---- emit top-layer hidden of para-1 at original time t ----------------
    for (int idx = tid; idx < 2048; idx += 512) {
      int b = idx >> 8, h = idx & 255;
      cat[((size_t)(b * TLEN + t)) * DDIM + sidx * HDIM + h] =
          sH[((0 * 2 + 1) * 16 + b) * HDIM + h];
    }
    __syncthreads();
  }  // timestep loop
}

// ---------------------------------------------------------------------------
// Host-side orchestration
// ---------------------------------------------------------------------------
extern "C" void kernel_launch(void* const* d_in, const int* in_sizes, int n_in,
                              void* d_out, int out_size, void* d_ws,
                              size_t ws_size, hipStream_t stream) {
  (void)in_sizes; (void)n_in; (void)out_size; (void)ws_size;
  const float* x = (const float*)d_in[0];
  const float* Wih0 = (const float*)d_in[1];
  const float* Whh0 = (const float*)d_in[2];
  const float* bih0 = (const float*)d_in[3];
  const float* bhh0 = (const float*)d_in[4];
  const float* Wih1 = (const float*)d_in[5];
  const float* Whh1 = (const float*)d_in[6];
  const float* bih1 = (const float*)d_in[7];
  const float* bhh1 = (const float*)d_in[8];
  const float* h0 = (const float*)d_in[9];
  const float* gateW = (const float*)d_in[10];
  const float* gate_b = (const float*)d_in[11];
  const float* linW = (const float*)d_in[12];
  const float* lin_b = (const float*)d_in[13];
  float* out = (float*)d_out;

  // Workspace layout (bytes)
  char* ws = (char*)d_ws;
  constexpr size_t MROWS = (size_t)TLEN * BATCH;  // 16384
  size_t off = 0;
  _Float16* x2h = (_Float16*)(ws + off);   off += MROWS * DDIM * 2;        // 25.2 MB
  _Float16* wih0h = (_Float16*)(ws + off); off += (size_t)6 * G3H * DDIM * 2;
  _Float16* whh0h = (_Float16*)(ws + off); off += (size_t)6 * G3H * HDIM * 2;
  _Float16* wih1h = (_Float16*)(ws + off); off += (size_t)6 * G3H * HDIM * 2;
  _Float16* whh1h = (_Float16*)(ws + off); off += (size_t)6 * G3H * HDIM * 2;
  _Float16* gateWh = (_Float16*)(ws + off); off += (size_t)HDIM * 512 * 2;
  _Float16* linWh = (_Float16*)(ws + off); off += (size_t)DDIM * DDIM * 2;
  float* gi0 = (float*)(ws + off);         off += (size_t)6 * MROWS * G3H * 4; // 302 MB
  float* cat = (float*)(ws + off);         off += MROWS * DDIM * 4;            // 50 MB
  _Float16* cath = (_Float16*)(ws + off);  off += MROWS * DDIM * 2;            // 25 MB

  auto blocks = [](size_t n) { return (unsigned)((n + 255) / 256); };

  // 1) x (B,T,D) -> f16, row = t*8+b
  xpose_to_f16<<<blocks(MROWS * DDIM), 256, 0, stream>>>(x, x2h);

  // 2) weights -> f16
  cvt_f32_to_f16<<<blocks((size_t)6 * G3H * DDIM), 256, 0, stream>>>(
      Wih0, wih0h, (size_t)6 * G3H * DDIM);
  cvt_f32_to_f16<<<blocks((size_t)6 * G3H * HDIM), 256, 0, stream>>>(
      Whh0, whh0h, (size_t)6 * G3H * HDIM);
  cvt_f32_to_f16<<<blocks((size_t)6 * G3H * HDIM), 256, 0, stream>>>(
      Wih1, wih1h, (size_t)6 * G3H * HDIM);
  cvt_f32_to_f16<<<blocks((size_t)6 * G3H * HDIM), 256, 0, stream>>>(
      Whh1, whh1h, (size_t)6 * G3H * HDIM);
  cvt_f32_to_f16<<<blocks((size_t)HDIM * 512), 256, 0, stream>>>(
      gateW, gateWh, (size_t)HDIM * 512);
  cvt_f32_to_f16<<<blocks((size_t)DDIM * DDIM), 256, 0, stream>>>(
      linW, linWh, (size_t)DDIM * DDIM);

  // 3) gi0[j] = x @ Wih0[j]^T + bih0[j]   (batched over 6 GRUs; N = 768)
  {
    dim3 grid((unsigned)(MROWS / 16), 1, 6);
    gemm_wmma_f16<<<grid, 256, 0, stream>>>(
        x2h, wih0h, bih0, gi0, DDIM, G3H,
        (long)G3H * DDIM, (long)G3H, (long)MROWS * G3H);
  }

  // 4) the three span scans (persistent blocks, ~246.5 KB dynamic LDS each)
  {
    const size_t smem = (size_t)(16384 + 16384 + 16 * GSTR + 16 * GSTR) *
                            sizeof(float) +
                        (size_t)16 * ASTR * sizeof(_Float16);
    span_scan<<<3, 512, smem, stream>>>(whh0h, wih1h, whh1h, gateWh, bhh0,
                                        bih1, bhh1, gate_b, h0, gi0, cat);
  }

  // 5) concat -> f16, then out = cat @ linW^T + lin_b   (N = 768)
  cvt_f32_to_f16<<<blocks(MROWS * DDIM), 256, 0, stream>>>(cat, cath,
                                                           MROWS * DDIM);
  {
    dim3 grid((unsigned)(MROWS / 16), 1, 1);
    gemm_wmma_f16<<<grid, 256, 0, stream>>>(cath, linWh, lin_b, out, DDIM,
                                            DDIM, 0, 0, 0);
  }
}
